// AttnDecoderRNN_1434519077451
// MI455X (gfx1250) — compile-verified, matrix operand order checked by reading, not run on hardware
//
#include <hip/hip_runtime.h>
#include <math.h>

// Problem constants (from reference)
#define VSZ 50257
#define ESZ 1024
#define HSZ 1024
#define OSZ 50257
#define BSZ 64
#define SSZ 512

typedef __attribute__((ext_vector_type(16))) __bf16 v16bf;
typedef __attribute__((ext_vector_type(8)))  float  v8f;

// ---------------------------------------------------------------------------
// WMMA GEMM: C[64,N] = A[64,K] @ B[K,N] + bias, fp32 in memory, bf16 MMA with
// fp32 accumulate. MT = M-tiles per wave (of 4 total), NT = N-tiles per wave.
// LDA/LDB/LDC/KK are compile-time so all hot-loop addressing folds into
// immediate offsets + one pointer bump per k-step.
// BOUNDS=false: every n-tile fully in range (no per-lane predication).
// BOUNDS=true: clamped addresses + cndmask zeroing for the ragged tail.
// Fragment layouts per CDNA5 ISA 7.12.2 (16-bit A 16x32, f32 C/D 16x16).
// ---------------------------------------------------------------------------
template <int MT, int NT, bool BOUNDS, int LDA, int LDB, int LDC, int KK>
__global__ void __launch_bounds__(256)
gemm_bias_wmma(const float* __restrict__ A,
               const float* __restrict__ Bm,
               const float* __restrict__ bias,
               float* __restrict__ C,
               int nbase, int N)
{
    constexpr int mSlots = 4 / MT;
    const int slot = blockIdx.x * 8 + (threadIdx.x >> 5);
    const int lane = threadIdx.x & 31;
    const int ng   = slot / mSlots;
    const int mt0  = (slot % mSlots) * MT;
    const int n0   = nbase + ng * (16 * NT);
    if (n0 >= N) return;                 // wave-uniform guard

    const int col = lane & 15;   // A row / B col / C col within tile
    const int hi  = lane >> 4;   // lane group -> +8 K offset

    v8f zf = {};
    v8f acc[MT][NT];
#pragma unroll
    for (int i = 0; i < MT; ++i)
#pragma unroll
        for (int j = 0; j < NT; ++j) acc[i][j] = zf;

    // ---- loop-invariant base pointers; only pointer bumps in the loop ----
    const float* bptr[NT];
    bool okn[NT];
#pragma unroll
    for (int j = 0; j < NT; ++j) {
        const int n = n0 + j * 16 + col;
        okn[j] = !BOUNDS || (n < N);
        const int nc = BOUNDS ? (okn[j] ? n : (N - 1)) : n;
        bptr[j] = Bm + (size_t)(8 * hi) * LDB + nc;
    }
    const float* aptr[MT];
#pragma unroll
    for (int i = 0; i < MT; ++i) {
        const int m = (mt0 + i) * 16 + col;
        aptr[i] = A + (size_t)m * LDA + 8 * hi;
    }
    const float* pf = Bm + (size_t)(32 + lane) * LDB + n0;

    for (int k0 = 0; k0 < KK; k0 += 32) {
        if (k0 + 32 < KK) {
            // prefetch next K-slab of the B stream (one cacheline per lane)
            __builtin_prefetch(pf, 0, 1);
        }
        pf += (size_t)32 * LDB;

        // ---- B fragments: lane holds one column, 16 K-values (imm offsets) ----
        v16bf bfrag[NT];
#pragma unroll
        for (int j = 0; j < NT; ++j) {
#pragma unroll
            for (int e = 0; e < 16; ++e) {
                const int koff = (e < 8) ? e : (e + 8);     // + 8*hi already in bptr
                float v = bptr[j][(size_t)koff * LDB];      // unconditional load
                if (BOUNDS) v = okn[j] ? v : 0.0f;          // plain v_cndmask
                bfrag[j][e] = (__bf16)v;
            }
            bptr[j] += (size_t)32 * LDB;
        }
        // ---- A fragments + MMA ----
#pragma unroll
        for (int i = 0; i < MT; ++i) {
            v16bf afrag;
#pragma unroll
            for (int e = 0; e < 8; ++e) {
                afrag[e]     = (__bf16)aptr[i][e];        // K = 8*hi + e
                afrag[e + 8] = (__bf16)aptr[i][e + 16];   // K = 8*hi + 16 + e
            }
            aptr[i] += 32;
#pragma unroll
            for (int j = 0; j < NT; ++j) {
                acc[i][j] = __builtin_amdgcn_wmma_f32_16x16x32_bf16(
                    false, afrag, false, bfrag[j], (short)0, acc[i][j],
                    false, false);
            }
        }
    }

    // ---- store: VGPR r -> M = r + 8*hi, N = col ----
#pragma unroll
    for (int j = 0; j < NT; ++j) {
        const int n = n0 + j * 16 + col;
        if (BOUNDS && n >= N) continue;
        const float bv = bias[n];
#pragma unroll
        for (int i = 0; i < MT; ++i) {
#pragma unroll
            for (int r = 0; r < 8; ++r) {
                const int m = (mt0 + i) * 16 + r + 8 * hi;
                C[(size_t)m * LDC + n] = acc[i][j][r] + bv;
            }
        }
    }
}

// ---------------------------------------------------------------------------
// X1[b, 0:1024] = emb[inputs[b]], X1[b, 1024:2048] = hidden[b]
// ---------------------------------------------------------------------------
__global__ void __launch_bounds__(256)
embed_concat_kernel(const int* __restrict__ inputs,
                    const float* __restrict__ emb,
                    const float* __restrict__ hidden,
                    float* __restrict__ X1)
{
    const int idx = blockIdx.x * 256 + threadIdx.x;   // 64*2048 total
    const int b = idx >> 11;
    const int c = idx & 2047;
    float v;
    if (c < ESZ) {
        v = emb[(size_t)inputs[b] * ESZ + c];
    } else {
        v = hidden[(size_t)b * HSZ + (c - ESZ)];
    }
    X1[(size_t)b * (ESZ + HSZ) + c] = v;
}

// ---------------------------------------------------------------------------
// scores[b,s] = dot(q[b,:], enc[s,b,:]) — one wave per (b,s)
// ---------------------------------------------------------------------------
__global__ void __launch_bounds__(256)
scores_kernel(const float* __restrict__ q,
              const float* __restrict__ enc,
              float* __restrict__ scores)
{
    const int gid  = blockIdx.x * 8 + (threadIdx.x >> 5); // wave id, B*S total
    const int lane = threadIdx.x & 31;
    const int b = gid >> 9;          // /S
    const int s = gid & (SSZ - 1);
    const float* e  = enc + ((size_t)s * BSZ + b) * HSZ;
    const float* qq = q + (size_t)b * HSZ;
    float sum = 0.0f;
    for (int h = lane; h < HSZ; h += 32) sum += qq[h] * e[h];
#pragma unroll
    for (int off = 16; off > 0; off >>= 1) sum += __shfl_down(sum, off, 32);
    if (lane == 0) scores[(size_t)b * SSZ + s] = sum;
}

// ---------------------------------------------------------------------------
// masked softmax over S per row b (block = 512 threads, one per s)
// ---------------------------------------------------------------------------
__global__ void __launch_bounds__(512)
softmax_kernel(const float* __restrict__ scores,
               const int* __restrict__ lengths,
               float* __restrict__ attw)
{
    __shared__ float red[512];
    const int b = blockIdx.x;
    const int s = threadIdx.x;
    const int len = lengths[b];
    const bool valid = (s < len);
    const float v = valid ? scores[(size_t)b * SSZ + s] : -INFINITY;

    red[s] = v;
    __syncthreads();
    for (int off = 256; off > 0; off >>= 1) {
        if (s < off) red[s] = fmaxf(red[s], red[s + off]);
        __syncthreads();
    }
    const float m = red[0];
    __syncthreads();

    const float ev = valid ? expf(v - m) : 0.0f;
    red[s] = ev;
    __syncthreads();
    for (int off = 256; off > 0; off >>= 1) {
        if (s < off) red[s] += red[s + off];
        __syncthreads();
    }
    const float denom = red[0];
    attw[(size_t)b * SSZ + s] = ev / denom;
}

// ---------------------------------------------------------------------------
// Split-K context: part[(b*8+c), h] = sum_{s in chunk c} w[b,s]*enc[s,b,h]
// ---------------------------------------------------------------------------
__global__ void __launch_bounds__(1024)
ctx_partial_kernel(const float* __restrict__ attw,
                   const float* __restrict__ enc,
                   float* __restrict__ part)
{
    const int b = blockIdx.x >> 3;
    const int c = blockIdx.x & 7;
    const int h = threadIdx.x;
    float acc = 0.0f;
    const int s0 = c * (SSZ / 8);
    for (int s = s0; s < s0 + (SSZ / 8); ++s) {
        acc += attw[(size_t)b * SSZ + s] * enc[((size_t)s * BSZ + b) * HSZ + h];
    }
    part[((size_t)b * 8 + c) * HSZ + h] = acc;
}

// X2[b,0:1024] = ctx[b], X2[b,1024:2048] = hidden[b]
__global__ void __launch_bounds__(1024)
ctx_reduce_kernel(const float* __restrict__ part,
                  const float* __restrict__ hidden,
                  float* __restrict__ X2)
{
    const int b = blockIdx.x;
    const int h = threadIdx.x;
    float acc = 0.0f;
#pragma unroll
    for (int c = 0; c < 8; ++c) acc += part[((size_t)b * 8 + c) * HSZ + h];
    X2[(size_t)b * (2 * HSZ) + h] = acc;
    X2[(size_t)b * (2 * HSZ) + HSZ + h] = hidden[(size_t)b * HSZ + h];
}

// ---------------------------------------------------------------------------
// GRU elementwise (PyTorch gate order r,z,n); writes h_new to ws and to the
// hidden_out slice of d_out.
// ---------------------------------------------------------------------------
__device__ __forceinline__ float sigmoidf_(float x) {
    return 1.0f / (1.0f + expf(-x));
}

__global__ void __launch_bounds__(256)
gru_kernel(const float* __restrict__ gi,
           const float* __restrict__ gh,
           const float* __restrict__ hcomb,
           float* __restrict__ hnew_ws,
           float* __restrict__ hnew_out)
{
    const int idx = blockIdx.x * 256 + threadIdx.x;   // 64*1024
    const int b = idx >> 10;
    const int j = idx & 1023;
    const size_t g = (size_t)b * (3 * HSZ) + j;
    const float i_r = gi[g], i_z = gi[g + HSZ], i_n = gi[g + 2 * HSZ];
    const float h_r = gh[g], h_z = gh[g + HSZ], h_n = gh[g + 2 * HSZ];
    const float r = sigmoidf_(i_r + h_r);
    const float z = sigmoidf_(i_z + h_z);
    const float n = tanhf(i_n + r * h_n);
    const float h = hcomb[(size_t)b * HSZ + j];
    const float hn = (1.0f - z) * n + z * h;
    hnew_ws[(size_t)b * HSZ + j] = hn;
    hnew_out[(size_t)b * HSZ + j] = hn;
}

// ---------------------------------------------------------------------------
// In-place row-wise log-softmax over O=50257 (block per row, 1024 threads)
// ---------------------------------------------------------------------------
__global__ void __launch_bounds__(1024)
log_softmax_kernel(float* __restrict__ out)
{
    __shared__ float red[1024];
    const int b = blockIdx.x;
    const int t = threadIdx.x;
    float* row = out + (size_t)b * OSZ;

    float m = -INFINITY;
    for (int i = t; i < OSZ; i += 1024) m = fmaxf(m, row[i]);
    red[t] = m;
    __syncthreads();
    for (int off = 512; off > 0; off >>= 1) {
        if (t < off) red[t] = fmaxf(red[t], red[t + off]);
        __syncthreads();
    }
    m = red[0];
    __syncthreads();

    float sum = 0.0f;
    for (int i = t; i < OSZ; i += 1024) sum += expf(row[i] - m);
    red[t] = sum;
    __syncthreads();
    for (int off = 512; off > 0; off >>= 1) {
        if (t < off) red[t] += red[t + off];
        __syncthreads();
    }
    const float lse = m + logf(red[0]);

    for (int i = t; i < OSZ; i += 1024) row[i] = row[i] - lse;
}

// ---------------------------------------------------------------------------
extern "C" void kernel_launch(void* const* d_in, const int* in_sizes, int n_in,
                              void* d_out, int out_size, void* d_ws, size_t ws_size,
                              hipStream_t stream)
{
    const int*   inputs  = (const int*)d_in[0];
    const float* hidden  = (const float*)d_in[1];
    const float* enc     = (const float*)d_in[2];
    const int*   enc_len = (const int*)d_in[3];
    const float* emb     = (const float*)d_in[4];
    const float* W_red   = (const float*)d_in[5];
    const float* b_red   = (const float*)d_in[6];
    const float* W_comb  = (const float*)d_in[7];
    const float* b_comb  = (const float*)d_in[8];
    const float* W_ih    = (const float*)d_in[9];
    const float* b_ih    = (const float*)d_in[10];
    const float* W_hh    = (const float*)d_in[11];
    const float* b_hh    = (const float*)d_in[12];
    const float* W_out   = (const float*)d_in[13];
    const float* b_out   = (const float*)d_in[14];
    float* out = (float*)d_out;

    float* ws = (float*)d_ws;
    float* X1   = ws;                      // 64*2048
    float* X2   = X1 + BSZ * 2048;         // 64*2048
    float* q    = X2 + BSZ * 2048;         // 64*1024
    float* sc   = q  + BSZ * HSZ;          // 64*512
    float* aw   = sc + BSZ * SSZ;          // 64*512
    float* nh   = aw + BSZ * SSZ;          // 64*1024
    float* gi   = nh + BSZ * HSZ;          // 64*3072
    float* gh   = gi + BSZ * 3 * HSZ;      // 64*3072
    float* hn   = gh + BSZ * 3 * HSZ;      // 64*1024
    float* part = hn + BSZ * HSZ;          // 64*8*1024

    // 1) X1 = [emb[inputs] | hidden]
    embed_concat_kernel<<<(BSZ * 2048) / 256, 256, 0, stream>>>(inputs, emb, hidden, X1);

    // 2) q = X1 @ W_red + b_red   [64,2048]x[2048,1024]
    {
        int slots = (HSZ / 16) * 4;                        // MT=1 -> 4 m-slots per n-tile
        gemm_bias_wmma<1, 1, false, 2048, HSZ, HSZ, 2048>
            <<<(slots + 7) / 8, 256, 0, stream>>>(X1, W_red, b_red, q, 0, HSZ);
    }

    // 3) scores[b,s] = q[b]·enc[s,b]
    scores_kernel<<<(BSZ * SSZ) / 8, 256, 0, stream>>>(q, enc, sc);

    // 4) masked softmax over S
    softmax_kernel<<<BSZ, SSZ, 0, stream>>>(sc, enc_len, aw);

    // 5) context (split-K) -> X2 = [ctx | hidden]
    ctx_partial_kernel<<<BSZ * 8, HSZ, 0, stream>>>(aw, enc, part);
    ctx_reduce_kernel<<<BSZ, HSZ, 0, stream>>>(part, hidden, X2);

    // 6) new_hidden = X2 @ W_comb + b_comb
    {
        int slots = (HSZ / 16) * 4;
        gemm_bias_wmma<1, 1, false, 2048, HSZ, HSZ, 2048>
            <<<(slots + 7) / 8, 256, 0, stream>>>(X2, W_comb, b_comb, nh, 0, HSZ);
    }

    // 7) gi = embedded @ W_ih + b_ih  (embedded = X1 cols [0,1024), lda=2048)
    //    gh = new_hidden @ W_hh + b_hh
    {
        int slots = ((3 * HSZ) / 16) * 4;
        gemm_bias_wmma<1, 1, false, 2048, 3 * HSZ, 3 * HSZ, HSZ>
            <<<(slots + 7) / 8, 256, 0, stream>>>(X1, W_ih, b_ih, gi, 0, 3 * HSZ);
        gemm_bias_wmma<1, 1, false, HSZ, 3 * HSZ, 3 * HSZ, HSZ>
            <<<(slots + 7) / 8, 256, 0, stream>>>(nh, W_hh, b_hh, gh, 0, 3 * HSZ);
    }

    // 8) GRU elementwise -> h_new (ws + hidden_out slice of d_out)
    gru_kernel<<<(BSZ * HSZ) / 256, 256, 0, stream>>>(
        gi, gh, nh, hn, out + (size_t)BSZ * OSZ);

    // 9) logits = h_new @ W_out + b_out  (big streaming GEMM)
    //    Bulk: full 32-column groups, zero predication. Tail: 17 ragged cols.
    {
        const int Nbulk = (OSZ / 32) * 32;                 // 50240
        int ngroups = Nbulk / 32;                          // NT=2 -> 32 cols/wave
        gemm_bias_wmma<4, 2, false, HSZ, OSZ, OSZ, HSZ>
            <<<(ngroups + 7) / 8, 256, 0, stream>>>(hn, W_out, b_out, out, 0, Nbulk);
        gemm_bias_wmma<4, 1, true, HSZ, OSZ, OSZ, HSZ>
            <<<1, 256, 0, stream>>>(hn, W_out, b_out, out, Nbulk, OSZ);
    }

    // 10) in-place log-softmax over O
    log_softmax_kernel<<<BSZ, 1024, 0, stream>>>(out);
}
